// ProtBertGCN_82841329205683
// MI455X (gfx1250) — matrix operand order repaired and checked by reading, not compiled, not run on hardware
//
#include <hip/hip_runtime.h>
#include <hip/hip_bf16.h>

typedef __bf16 bf16;
typedef __attribute__((ext_vector_type(16))) __bf16 v16bf;
typedef __attribute__((ext_vector_type(8)))  __bf16 v8bf;
typedef __attribute__((ext_vector_type(8)))  float  v8f;

#define WMMA_BF16(a, b, c) __builtin_amdgcn_wmma_f32_16x16x32_bf16( \
        false, (a), false, (b), (short)0, (c), false, false)

// ---------------------------------------------------------------------------
// Small utility kernels
// ---------------------------------------------------------------------------
__global__ void fill_f32_kernel(float* p, long n, float v) {
    long i = (long)blockIdx.x * blockDim.x + threadIdx.x;
    if (i < n) p[i] = v;
}

__global__ void degree_kernel(const int* __restrict__ dst, long E, float* __restrict__ deg) {
    long e = (long)blockIdx.x * blockDim.x + threadIdx.x;
    if (e < E) atomicAdd(&deg[dst[e]], 1.0f);
}

__global__ void rsqrt_kernel(float* __restrict__ deg, long n) {
    long i = (long)blockIdx.x * blockDim.x + threadIdx.x;
    if (i < n) {
        float d = deg[i];
        deg[i] = (d > 0.0f) ? rsqrtf(d) : 0.0f;   // deg >= 1 due to self loop
    }
}

// h_bf[node][c] = bf16(emb[x[node]][c]) ; leading dim = C
__global__ void embed_kernel(const int* __restrict__ x, const float* __restrict__ emb,
                             bf16* __restrict__ hbf, long N, int C) {
    long idx = (long)blockIdx.x * blockDim.x + threadIdx.x;
    if (idx < N * (long)C) {
        long node = idx / C;
        int  c    = (int)(idx - node * C);
        hbf[idx] = (bf16)emb[(size_t)x[node] * C + c];
    }
}

// Wt[n][k] = bf16(W[k][n]) for n < Nout, zero-padded up to NP rows.
// (W is K x Nout row-major; Wt is NP x K row-major)
__global__ void convert_wt_kernel(const float* __restrict__ W, bf16* __restrict__ Wt,
                                  int K, int Nout, int NP) {
    long idx = (long)blockIdx.x * blockDim.x + threadIdx.x;
    if (idx < (long)NP * K) {
        long n = idx / K;
        int  k = (int)(idx - n * K);
        float v = (n < (long)Nout) ? W[(size_t)k * Nout + n] : 0.0f;
        Wt[idx] = (bf16)v;
    }
}

// ---------------------------------------------------------------------------
// WMMA bf16 GEMM, 32x64 output tile per wave:
//   2 A fragments x 4 B fragments -> 8 independent accumulators,
//   8 v_wmma_f32_16x16x32_bf16 per 32-wide k-step, 12 b128 loads.
//   A:  M x K bf16 row-major (leading dim lda)
//   Bt: NP x K bf16 row-major (= B^T, zero-padded to NP multiple of 64)
// Fused epilogue:
//   dinv != null:  Cmain = acc (hw) ; Cagg = acc*dinv[row]^2 + bias[col]
//   dinv == null:  Cmain = acc + bias[col]   (classifier output)
// Lane layouts used here: A 16-bit 16x32 (lane<16: K{0..15},
// lane>=16: K{16..31}); B lane = column, halves K 0..15 / 16..31
// (contiguous 32B from transposed weights); D: VGPR r -> M=r+8*(lane>=16),
// N=lane&15.
// ---------------------------------------------------------------------------
__global__ void wmma_gemm_bf16_kernel(const bf16* __restrict__ A, int lda,
                                      const bf16* __restrict__ Bt, int NP,
                                      float* __restrict__ Cmain,
                                      float* __restrict__ Cagg,
                                      const float* __restrict__ bias,
                                      const float* __restrict__ dinv,
                                      int M, int Nout, int K) {
    int wave = blockIdx.x * (blockDim.x >> 5) + (threadIdx.x >> 5);
    int lane = threadIdx.x & 31;

    int tilesN = NP >> 6;                 // 64-wide col tiles
    int tilesM = (M + 31) >> 5;           // 32-high row tiles
    if (wave >= tilesM * tilesN) return;

    int tm = wave / tilesN;
    int tn = wave - tm * tilesN;
    int m0 = tm << 5;
    int n0 = tn << 6;

    int r16  = lane & 15;
    int half = lane >> 4;

    int ar0 = m0 + r16;       if (ar0 >= M) ar0 = M - 1;   // stores are guarded
    int ar1 = m0 + 16 + r16;  if (ar1 >= M) ar1 = M - 1;
    const bf16* arow0 = A + (size_t)ar0 * lda;
    const bf16* arow1 = A + (size_t)ar1 * lda;

    // hoisted B row base pointers (one per 16-wide col tile)
    const bf16* brow0 = Bt + (size_t)(n0 + r16) * K + half * 16;
    const bf16* brow1 = brow0 + (size_t)16 * K;
    const bf16* brow2 = brow0 + (size_t)32 * K;
    const bf16* brow3 = brow0 + (size_t)48 * K;

    v8f acc[2][4] = {};
    for (int k0 = 0; k0 < K; k0 += 32) {
        union { v8bf h[2]; v16bf w; } ua0, ua1;
        ua0.h[0] = *(const v8bf*)(arow0 + k0 + half * 16);
        ua0.h[1] = *(const v8bf*)(arow0 + k0 + half * 16 + 8);
        ua1.h[0] = *(const v8bf*)(arow1 + k0 + half * 16);
        ua1.h[1] = *(const v8bf*)(arow1 + k0 + half * 16 + 8);
        v16bf a0 = ua0.w;
        v16bf a1 = ua1.w;

        v16bf b0 = *(const v16bf*)(brow0 + k0);
        v16bf b1 = *(const v16bf*)(brow1 + k0);
        v16bf b2 = *(const v16bf*)(brow2 + k0);
        v16bf b3 = *(const v16bf*)(brow3 + k0);

        acc[0][0] = WMMA_BF16(a0, b0, acc[0][0]);
        acc[1][0] = WMMA_BF16(a1, b0, acc[1][0]);
        acc[0][1] = WMMA_BF16(a0, b1, acc[0][1]);
        acc[1][1] = WMMA_BF16(a1, b1, acc[1][1]);
        acc[0][2] = WMMA_BF16(a0, b2, acc[0][2]);
        acc[1][2] = WMMA_BF16(a1, b2, acc[1][2]);
        acc[0][3] = WMMA_BF16(a0, b3, acc[0][3]);
        acc[1][3] = WMMA_BF16(a1, b3, acc[1][3]);
    }

    if (dinv) {
        // GCN layer: write hw and agg = hw*dinv^2 + bias
#pragma unroll
        for (int i = 0; i < 2; ++i) {
            int rowBase = m0 + 16 * i + half * 8;
#pragma unroll
            for (int r = 0; r < 8; ++r) {
                int row = rowBase + r;
                if (row >= M) continue;
                float di = dinv[row];
                float sl = di * di;
                size_t base = (size_t)row * Nout;
#pragma unroll
                for (int j = 0; j < 4; ++j) {
                    int col = n0 + 16 * j + r16;
                    if (col < Nout) {
                        float v = acc[i][j][r];
                        Cmain[base + col] = v;
                        Cagg [base + col] = v * sl + bias[col];
                    }
                }
            }
        }
    } else {
        // classifier: out = acc + bias
#pragma unroll
        for (int i = 0; i < 2; ++i) {
            int rowBase = m0 + 16 * i + half * 8;
#pragma unroll
            for (int r = 0; r < 8; ++r) {
                int row = rowBase + r;
                if (row >= M) continue;
                size_t base = (size_t)row * Nout;
#pragma unroll
                for (int j = 0; j < 4; ++j) {
                    int col = n0 + 16 * j + r16;
                    if (col < Nout) Cmain[base + col] = acc[i][j][r] + bias[col];
                }
            }
        }
    }
}

// ---------------------------------------------------------------------------
// Edge scatter: one thread per (edge, 4-feature chunk): agg[dst] += hw[src]*norm
// ---------------------------------------------------------------------------
__global__ void scatter_edges_kernel(const int* __restrict__ src, const int* __restrict__ dst,
                                     const float* __restrict__ hw, const float* __restrict__ dinv,
                                     float* __restrict__ agg, long E, int C4 /* C/4 */, int C) {
    long idx = (long)blockIdx.x * blockDim.x + threadIdx.x;
    if (idx >= E * (long)C4) return;
    long e     = idx / C4;
    int  chunk = (int)(idx - e * C4);
    int  s = src[e], d = dst[e];
    float norm = dinv[s] * dinv[d];
    const float4 v = ((const float4*)(hw + (size_t)s * C))[chunk];
    float* out = agg + (size_t)d * C + (size_t)chunk * 4;
    atomicAdd(out + 0, v.x * norm);
    atomicAdd(out + 1, v.y * norm);
    atomicAdd(out + 2, v.z * norm);
    atomicAdd(out + 3, v.w * norm);
}

// in-place ReLU on agg, plus bf16 copy for the next GEMM's A operand
__global__ void relu_bf16_kernel(float* __restrict__ agg, bf16* __restrict__ hbf,
                                 long N, int C) {
    long idx = (long)blockIdx.x * blockDim.x + threadIdx.x;
    if (idx < N * (long)C) {
        float r = agg[idx];
        r = (r > 0.0f) ? r : 0.0f;
        agg[idx] = r;
        hbf[idx] = (bf16)r;
    }
}

// ---------------------------------------------------------------------------
// Pooling
// ---------------------------------------------------------------------------
__global__ void count_kernel(const int* __restrict__ batch, float* __restrict__ cnt, long N) {
    long i = (long)blockIdx.x * blockDim.x + threadIdx.x;
    if (i < N) atomicAdd(&cnt[batch[i]], 1.0f);
}

__global__ void pool_sum_kernel(const float* __restrict__ h, const int* __restrict__ batch,
                                float* __restrict__ psum, long N, int C) {
    long idx = (long)blockIdx.x * blockDim.x + threadIdx.x;
    if (idx < N * (long)C) {
        long i = idx / C;
        int  c = (int)(idx - i * C);
        atomicAdd(&psum[(size_t)batch[i] * C + c], h[idx]);
    }
}

__global__ void pool_div_kernel(const float* __restrict__ psum, const float* __restrict__ cnt,
                                bf16* __restrict__ pbf, int G, int C) {
    int idx = blockIdx.x * blockDim.x + threadIdx.x;
    if (idx < G * C) {
        int g = idx / C;
        float d = cnt[g];
        d = (d > 1.0f) ? d : 1.0f;
        pbf[idx] = (bf16)(psum[idx] / d);
    }
}

// ---------------------------------------------------------------------------
// Host launch
// ---------------------------------------------------------------------------
static inline unsigned blocks_for(long n, int t) { return (unsigned)((n + t - 1) / t); }

extern "C" void kernel_launch(void* const* d_in, const int* in_sizes, int n_in,
                              void* d_out, int out_size, void* d_ws, size_t ws_size,
                              hipStream_t stream) {
    const int*   x     = (const int*)d_in[0];
    const int*   eidx  = (const int*)d_in[1];
    const int*   batch = (const int*)d_in[2];
    const float* emb   = (const float*)d_in[3];
    const float* W1    = (const float*)d_in[4];
    const float* b1    = (const float*)d_in[5];
    const float* W2    = (const float*)d_in[6];
    const float* b2    = (const float*)d_in[7];
    const float* Wc    = (const float*)d_in[8];
    const float* bc    = (const float*)d_in[9];
    float* out = (float*)d_out;

    const long N  = in_sizes[0];               // 100000 nodes
    const long E  = in_sizes[1] / 2;           // 1.6M edges
    const int  H1 = in_sizes[5];               // 256
    const int  K1 = in_sizes[4] / H1;          // 128
    const int  H2 = in_sizes[7];               // 256
    const int  NC = in_sizes[9];               // 30954
    const int  KC = in_sizes[8] / NC;          // 256
    const int  G  = 64;                        // NUM_GRAPHS

    const int H1P = (H1 + 63) & ~63;           // padded col counts (mult of 64)
    const int H2P = (H2 + 63) & ~63;
    const int NCP = (NC + 63) & ~63;

    const int* esrc = eidx;
    const int* edst = eidx + E;

    // bump allocator over workspace
    char* wsp = (char*)d_ws;
    auto alloc = [&](size_t bytes) -> void* {
        void* p = (void*)wsp;
        wsp += (bytes + 255) & ~(size_t)255;
        return p;
    };
    float* deg  = (float*)alloc((size_t)N * 4);            // -> dinv in place
    bf16*  hbf  = (bf16*) alloc((size_t)N * H2 * 2);       // GEMM A operand
    float* hw   = (float*)alloc((size_t)N * H2 * 4);       // GEMM output (messages)
    float* agg  = (float*)alloc((size_t)N * H2 * 4);       // aggregation buffer
    bf16*  w1t  = (bf16*) alloc((size_t)H1P * K1 * 2);     // W1^T bf16 (padded)
    bf16*  w2t  = (bf16*) alloc((size_t)H2P * H1 * 2);     // W2^T bf16 (padded)
    bf16*  wct  = (bf16*) alloc((size_t)NCP * KC * 2);     // Wc^T bf16 (padded)
    float* psum = (float*)alloc((size_t)G * H2 * 4);
    float* cnt  = (float*)alloc((size_t)G * 4);
    bf16*  pbf  = (bf16*) alloc((size_t)G * KC * 2);

    const int T = 256;
    const int wavesPerBlock = T >> 5;

    // degrees (init to 1.0 = self loop), then dinv = rsqrt(deg)
    fill_f32_kernel<<<blocks_for(N, T), T, 0, stream>>>(deg, N, 1.0f);
    degree_kernel<<<blocks_for(E, T), T, 0, stream>>>(edst, E, deg);
    rsqrt_kernel<<<blocks_for(N, T), T, 0, stream>>>(deg, N);
    float* dinv = deg;

    // embedding lookup -> bf16 A operand (ld = K1)
    embed_kernel<<<blocks_for(N * (long)K1, T), T, 0, stream>>>(x, emb, hbf, N, K1);

    // weight conversions (transposed + zero padded)
    convert_wt_kernel<<<blocks_for((long)K1 * H1P, T), T, 0, stream>>>(W1, w1t, K1, H1, H1P);
    convert_wt_kernel<<<blocks_for((long)H1 * H2P, T), T, 0, stream>>>(W2, w2t, H1, H2, H2P);
    convert_wt_kernel<<<blocks_for((long)KC * NCP, T), T, 0, stream>>>(Wc, wct, KC, NC, NCP);

    // ---------------- layer 1 ----------------
    {
        long tiles = (long)((N + 31) >> 5) * (H1P >> 6);
        wmma_gemm_bf16_kernel<<<blocks_for(tiles, wavesPerBlock), T, 0, stream>>>(
            hbf, K1, w1t, H1P, hw, agg, b1, dinv, (int)N, H1, K1);
        scatter_edges_kernel<<<blocks_for(E * (long)(H1 / 4), T), T, 0, stream>>>(
            esrc, edst, hw, dinv, agg, E, H1 / 4, H1);
        relu_bf16_kernel<<<blocks_for(N * (long)H1, T), T, 0, stream>>>(agg, hbf, N, H1);
    }

    // ---------------- layer 2 ----------------
    {
        long tiles = (long)((N + 31) >> 5) * (H2P >> 6);
        wmma_gemm_bf16_kernel<<<blocks_for(tiles, wavesPerBlock), T, 0, stream>>>(
            hbf, H1, w2t, H2P, hw, agg, b2, dinv, (int)N, H2, H1);
        scatter_edges_kernel<<<blocks_for(E * (long)(H2 / 4), T), T, 0, stream>>>(
            esrc, edst, hw, dinv, agg, E, H2 / 4, H2);
        relu_bf16_kernel<<<blocks_for(N * (long)H2, T), T, 0, stream>>>(agg, hbf, N, H2);
    }

    // ---------------- mean pooling ----------------
    fill_f32_kernel<<<blocks_for((long)G * H2, T), T, 0, stream>>>(psum, (long)G * H2, 0.0f);
    fill_f32_kernel<<<1, G, 0, stream>>>(cnt, G, 0.0f);
    count_kernel<<<blocks_for(N, T), T, 0, stream>>>(batch, cnt, N);
    pool_sum_kernel<<<blocks_for(N * (long)H2, T), T, 0, stream>>>(agg, batch, psum, N, H2);
    pool_div_kernel<<<blocks_for((long)G * H2, T), T, 0, stream>>>(psum, cnt, pbf, G, H2);

    // ---------------- classifier: (64 x 256) @ (256 x 30954) + bc ----------------
    {
        long tiles = (long)((G + 31) >> 5) * (NCP >> 6);
        wmma_gemm_bf16_kernel<<<blocks_for(tiles, wavesPerBlock), T, 0, stream>>>(
            pbf, KC, wct, NCP, out, nullptr, bc, nullptr, G, NC, KC);
    }
}
